// Guide_46943992545591
// MI455X (gfx1250) — matrix operand order repaired and checked by reading, not compiled, not run on hardware
//
#include <hip/hip_runtime.h>
#include <hip/hip_bf16.h>
#include <math.h>

// ---------------------------------------------------------------------------
// Problem constants (match reference)
// ---------------------------------------------------------------------------
#define BB     4096      // batch B
#define SS     16        // seq S
#define NTOT   65536     // S*B
#define HID    256
#define G4     1024      // 4*HID
#define OBS_D  400
#define IN_D   432
#define LOG2PI_F 1.8378770664093454f

typedef _Float16 h8   __attribute__((ext_vector_type(8)));
typedef _Float16 v16h __attribute__((ext_vector_type(16)));
typedef float    v8f  __attribute__((ext_vector_type(8)));
typedef float    f4v  __attribute__((ext_vector_type(4)));

union V16U { v16h v; h8 h[2]; };

__device__ __forceinline__ v8f vzero8() {
    v8f v;
#pragma unroll
    for (int i = 0; i < 8; ++i) v[i] = 0.f;
    return v;
}

__device__ __forceinline__ h8 hzero8() {
    h8 v;
#pragma unroll
    for (int i = 0; i < 8; ++i) v[i] = (_Float16)0.f;
    return v;
}

__device__ __forceinline__ h8 pack_h8(f4v a, f4v b) {
    h8 r;
    r[0] = (_Float16)a[0]; r[1] = (_Float16)a[1]; r[2] = (_Float16)a[2]; r[3] = (_Float16)a[3];
    r[4] = (_Float16)b[0]; r[5] = (_Float16)b[1]; r[6] = (_Float16)b[2]; r[7] = (_Float16)b[3];
    return r;
}

__device__ __forceinline__ float sigmoidf_(float x) { return 1.f / (1.f + expf(-x)); }

// A-operand (16x32 f16), row-major matrix [row][k], leading dim ld (f16 units).
// Lane l<16: row=l, k = {k0..k0+7, k0+16..k0+23}; lane>=16: row=l-16, +8.
__device__ __forceinline__ v16h load_a16(const _Float16* mat, int ld, int row0, int k0, int lane) {
    int r = row0 + (lane & 15);
    int k = k0 + ((lane & 16) ? 8 : 0);
    const _Float16* p = mat + r * ld + k;
    V16U t;
    t.h[0] = *(const h8*)(p);
    t.h[1] = *(const h8*)(p + 16);
    return t.v;
}

// B-operand (32x16 f16) from matrix stored [n][k] row-major (i.e. W^T access).
// Lane l<16: n=n0+l, k = {k0..k0+7, k0+16..k0+23}; lane>=16: +8.
__device__ __forceinline__ v16h load_b16(const _Float16* mat, int ld, int n0, int k0, int lane) {
    int n = n0 + (lane & 15);
    int k = k0 + ((lane & 16) ? 8 : 0);
    const _Float16* p = mat + n * ld + k;
    V16U t;
    t.h[0] = *(const h8*)(p);
    t.h[1] = *(const h8*)(p + 16);
    return t.v;
}

// A-operand for the rp_emb extra K-tile: rows = rpe[nbase+m][0..15], K 16..31 zero.
__device__ __forceinline__ v16h load_a_rpe(const _Float16* rpe, int nbase, int lane) {
    int m = lane & 15;
    const _Float16* p = rpe + (size_t)(nbase + m) * 16 + ((lane & 16) ? 8 : 0);
    V16U t;
    t.h[0] = *(const h8*)(p);
    t.h[1] = hzero8();
    return t.v;
}

// B-operand for the rp_emb extra K-tile: wihmid[n][0..15] in k-rows 0..15, rows 16..31 zero.
__device__ __forceinline__ v16h load_b_mid(const _Float16* wihmid, int n0, int lane) {
    V16U t;
    {
        const _Float16* p = wihmid + (size_t)(n0 + (lane & 15)) * 16 + ((lane & 16) ? 8 : 0);
        t.h[0] = *(const h8*)(p);
        t.h[1] = hzero8();
    }
    return t.v;
}

// ---------------------------------------------------------------------------
// Kernel 1: tables + weight conversion
// ---------------------------------------------------------------------------
__global__ void k_tables(const float* __restrict__ W_ih, const float* __restrict__ W_hh,
                         const float* __restrict__ b_ih, const float* __restrict__ b_hh,
                         const float* __restrict__ addr_emb, const float* __restrict__ pid_emb,
                         const float* __restrict__ sid_emb,
                         float* __restrict__ addr_bias, float* __restrict__ tab_prog,
                         float* __restrict__ tab_shape,
                         _Float16* __restrict__ whh_h, _Float16* __restrict__ wihmid) {
    int nrow = blockIdx.x;       // 0..1023 (gate row)
    int t = threadIdx.x;         // 256
    whh_h[(size_t)nrow * 256 + t] = (_Float16)W_hh[(size_t)nrow * 256 + t];
    if (t < 16) wihmid[(size_t)nrow * 16 + t] = (_Float16)W_ih[(size_t)nrow * IN_D + 400 + t];
    if (t < 7) {
        float s = b_ih[nrow] + b_hh[nrow];
#pragma unroll
        for (int k = 0; k < 16; ++k) s += addr_emb[t * 16 + k] * W_ih[(size_t)nrow * IN_D + 416 + k];
        addr_bias[t * G4 + nrow] = s;
    } else if (t < 10) {
        int j = t - 7; float s = 0.f;
#pragma unroll
        for (int k = 0; k < 16; ++k) s += pid_emb[j * 16 + k] * W_ih[(size_t)nrow * IN_D + 400 + k];
        tab_prog[j * G4 + nrow] = s;
    } else if (t < 12) {
        int j = t - 10; float s = 0.f;
#pragma unroll
        for (int k = 0; k < 16; ++k) s += sid_emb[j * 16 + k] * W_ih[(size_t)nrow * IN_D + 400 + k];
        tab_shape[j * G4 + nrow] = s;
    }
}

// ---------------------------------------------------------------------------
// Kernel 2: obs_part[B][1024] = obs_e (B x 400) @ W_ih[:, :400].T   (WMMA f16)
// 128x128 output tile / workgroup, 8 waves.
// Staging is fully vectorized: each thread owns one half-row (16 elems) of the
// 128x32 K-tile; OBS_D=400 boundary aligns exactly to half-rows.
// ---------------------------------------------------------------------------
__launch_bounds__(256)
__global__ void k_obs_gemm(const float* __restrict__ obs, const float* __restrict__ W_ih,
                           float* __restrict__ obs_part) {
    __shared__ __align__(16) _Float16 Asb[128 * 32];
    __shared__ __align__(16) _Float16 Bsb[128 * 32];
    int m0 = blockIdx.x * 128;
    int n0 = blockIdx.y * 128;
    int tid = threadIdx.x, wave = tid >> 5, lane = tid & 31;
    int wr = wave & 3, wc = wave >> 2;

    int sr = tid >> 1;            // staging row 0..127
    int skq = (tid & 1) << 4;     // 0 or 16 (half-row)

    v8f acc[2][4];
#pragma unroll
    for (int mt = 0; mt < 2; ++mt)
#pragma unroll
        for (int nt = 0; nt < 4; ++nt) acc[mt][nt] = vzero8();

    for (int kt = 0; kt < 13; ++kt) {
        int k0 = kt * 32;
        h8 a0h, a1h, b0h, b1h;
        if (k0 + skq < OBS_D) {   // whole half-row valid (400 is a multiple of 16)
            const f4v* ga = (const f4v*)(obs + (size_t)(m0 + sr) * OBS_D + k0 + skq);
            const f4v* gb = (const f4v*)(W_ih + (size_t)(n0 + sr) * IN_D + k0 + skq);
            f4v a0 = ga[0], a1 = ga[1], a2 = ga[2], a3 = ga[3];
            f4v b0 = gb[0], b1 = gb[1], b2 = gb[2], b3 = gb[3];
            a0h = pack_h8(a0, a1); a1h = pack_h8(a2, a3);
            b0h = pack_h8(b0, b1); b1h = pack_h8(b2, b3);
        } else {
            a0h = hzero8(); a1h = hzero8(); b0h = hzero8(); b1h = hzero8();
        }
        *(h8*)(Asb + sr * 32 + skq)     = a0h;
        *(h8*)(Asb + sr * 32 + skq + 8) = a1h;
        *(h8*)(Bsb + sr * 32 + skq)     = b0h;
        *(h8*)(Bsb + sr * 32 + skq + 8) = b1h;
        __syncthreads();
        v16h a0 = load_a16(Asb, 32, wr * 32 + 0,  0, lane);
        v16h a1 = load_a16(Asb, 32, wr * 32 + 16, 0, lane);
#pragma unroll
        for (int nt = 0; nt < 4; ++nt) {
            v16h bt = load_b16(Bsb, 32, wc * 64 + nt * 16, 0, lane);
            acc[0][nt] = __builtin_amdgcn_wmma_f32_16x16x32_f16(false, a0, false, bt, (short)0, acc[0][nt], false, false);
            acc[1][nt] = __builtin_amdgcn_wmma_f32_16x16x32_f16(false, a1, false, bt, (short)0, acc[1][nt], false, false);
        }
        __syncthreads();
    }
#pragma unroll
    for (int mt = 0; mt < 2; ++mt)
#pragma unroll
        for (int nt = 0; nt < 4; ++nt)
#pragma unroll
            for (int r = 0; r < 8; ++r) {
                int row = m0 + wr * 32 + mt * 16 + r + ((lane & 16) ? 8 : 0);
                int col = n0 + wc * 64 + nt * 16 + (lane & 15);
                obs_part[(size_t)row * G4 + col] = acc[mt][nt][r];
            }
}

// ---------------------------------------------------------------------------
// Kernel 3: step-1 LSTM per b (h0=c0=0) + program log-softmax per b
// ---------------------------------------------------------------------------
__global__ void k_h1(const float* __restrict__ obs_part, const float* __restrict__ addr_bias,
                     const float* __restrict__ Wp, const float* __restrict__ bp,
                     float* __restrict__ c1, _Float16* __restrict__ h1h,
                     float* __restrict__ plog) {
    int b = blockIdx.x, u = threadIdx.x;
    const float* op = obs_part + (size_t)b * G4;
    float gi = op[u]       + addr_bias[u];
    float gg = op[512 + u] + addr_bias[512 + u];
    float go = op[768 + u] + addr_bias[768 + u];
    float c = sigmoidf_(gi) * tanhf(gg);            // f-gate * c0 = 0
    float h = sigmoidf_(go) * tanhf(c);
    c1[(size_t)b * HID + u] = c;
    h1h[(size_t)b * HID + u] = (_Float16)h;

    __shared__ float red[256];
    __shared__ float lg[3];
    for (int j = 0; j < 3; ++j) {
        red[u] = h * Wp[j * HID + u];
        __syncthreads();
        for (int off = 128; off > 0; off >>= 1) {
            if (u < off) red[u] += red[u + off];
            __syncthreads();
        }
        if (u == 0) lg[j] = red[0] + bp[j];
        __syncthreads();
    }
    if (u == 0) {
        float m = fmaxf(lg[0], fmaxf(lg[1], lg[2]));
        float lse = m + logf(expf(lg[0] - m) + expf(lg[1] - m) + expf(lg[2] - m));
        plog[b * 3 + 0] = lg[0] - lse;
        plog[b * 3 + 1] = lg[1] - lse;
        plog[b * 3 + 2] = lg[2] - lse;
    }
}

// ---------------------------------------------------------------------------
// Kernel 4: per-row MLP: rpos[:,0] -> 100 -> 100 -> 16 (rp_emb, f16)
// ---------------------------------------------------------------------------
__global__ void k_mlp(const float* __restrict__ raw_pos,
                      const float* __restrict__ W1, const float* __restrict__ b1,
                      const float* __restrict__ W2, const float* __restrict__ b2,
                      const float* __restrict__ W3, const float* __restrict__ b3,
                      _Float16* __restrict__ rpe) {
    int n = blockIdx.x, t = threadIdx.x;   // 128 threads
    __shared__ __align__(16) float hh1[100];
    __shared__ __align__(16) float hh2[100];
    float x0 = raw_pos[(size_t)n * 4 + 0];
    float y0 = raw_pos[(size_t)n * 4 + 1];
    if (t < 100) hh1[t] = fmaxf(0.f, x0 * W1[t * 2] + y0 * W1[t * 2 + 1] + b1[t]);
    __syncthreads();
    if (t < 100) {
        float s = b2[t];
        const f4v* w = (const f4v*)(W2 + t * 100);
        const f4v* hv = (const f4v*)hh1;
#pragma unroll 5
        for (int k = 0; k < 25; ++k) {
            f4v wv = w[k], hx = hv[k];
            s += wv[0] * hx[0] + wv[1] * hx[1] + wv[2] * hx[2] + wv[3] * hx[3];
        }
        hh2[t] = fmaxf(0.f, s);
    }
    __syncthreads();
    if (t < 16) {
        float s = b3[t];
        const f4v* w = (const f4v*)(W3 + t * 100);
        const f4v* hv = (const f4v*)hh2;
#pragma unroll 5
        for (int k = 0; k < 25; ++k) {
            f4v wv = w[k], hx = hv[k];
            s += wv[0] * hx[0] + wv[1] * hx[1] + wv[2] * hx[2] + wv[3] * hx[3];
        }
        rpe[(size_t)n * 16 + t] = (_Float16)s;
    }
}

// ---------------------------------------------------------------------------
// Main kernel helpers
// ---------------------------------------------------------------------------
// One LSTM step for 32 rows. 8 waves; wave w owns hidden units [w*32, w*32+32)
// of all four gate quadrants -> nonlinearity is wave-local, c stays in VGPRs.
__device__ __forceinline__ void do_step(
    _Float16* hL, v8f (&creg)[2][2],
    const _Float16* __restrict__ whh_h, const float* __restrict__ obs_part,
    const float* __restrict__ ab,                 // addr_bias row (1024)
    const float* __restrict__ tab,                // prev-emb table or nullptr
    const int* selL,                              // per-row selector (LDS) or nullptr
    const _Float16* __restrict__ rpe,             // rp_emb (step F) or nullptr
    const _Float16* __restrict__ wihmid,
    const float* __restrict__ c1, bool init_c,
    int b0, int n0, int wave, int lane) {
    int ub = wave * 32;

    // Warm the weight stream for this wave's gate rows (global_prefetch_b8).
#pragma unroll
    for (int q = 0; q < 4; ++q)
        __builtin_prefetch((const void*)(whh_h + (size_t)(q * 256 + ub + lane) * 256), 0, 2);

    v8f acc[2][4][2];
#pragma unroll
    for (int mt = 0; mt < 2; ++mt)
#pragma unroll
        for (int q = 0; q < 4; ++q)
#pragma unroll
            for (int u2 = 0; u2 < 2; ++u2) acc[mt][q][u2] = vzero8();

    if (init_c) {
#pragma unroll
        for (int mt = 0; mt < 2; ++mt)
#pragma unroll
            for (int u2 = 0; u2 < 2; ++u2)
#pragma unroll
                for (int r = 0; r < 8; ++r) {
                    int row = mt * 16 + r + ((lane & 16) ? 8 : 0);
                    int cu = ub + u2 * 16 + (lane & 15);
                    creg[mt][u2][r] = c1[(size_t)(b0 + row) * HID + cu];
                }
    }

#pragma unroll 2
    for (int kt = 0; kt < 8; ++kt) {
        int k0 = kt * 32;
        v16h a0 = load_a16(hL, 256, 0,  k0, lane);
        v16h a1 = load_a16(hL, 256, 16, k0, lane);
#pragma unroll
        for (int q = 0; q < 4; ++q)
#pragma unroll
            for (int u2 = 0; u2 < 2; ++u2) {
                v16h bt = load_b16(whh_h, 256, q * 256 + ub + u2 * 16, k0, lane);
                acc[0][q][u2] = __builtin_amdgcn_wmma_f32_16x16x32_f16(false, a0, false, bt, (short)0, acc[0][q][u2], false, false);
                acc[1][q][u2] = __builtin_amdgcn_wmma_f32_16x16x32_f16(false, a1, false, bt, (short)0, acc[1][q][u2], false, false);
            }
    }

    if (rpe != nullptr) {   // rp_emb @ W_ih_mid.T as one zero-padded K-tile
        v16h a0 = load_a_rpe(rpe, n0 + 0,  lane);
        v16h a1 = load_a_rpe(rpe, n0 + 16, lane);
#pragma unroll
        for (int q = 0; q < 4; ++q)
#pragma unroll
            for (int u2 = 0; u2 < 2; ++u2) {
                v16h bt = load_b_mid(wihmid, q * 256 + ub + u2 * 16, lane);
                acc[0][q][u2] = __builtin_amdgcn_wmma_f32_16x16x32_f16(false, a0, false, bt, (short)0, acc[0][q][u2], false, false);
                acc[1][q][u2] = __builtin_amdgcn_wmma_f32_16x16x32_f16(false, a1, false, bt, (short)0, acc[1][q][u2], false, false);
            }
    }

    __syncthreads();   // all waves done reading hL before we overwrite it

#pragma unroll
    for (int mt = 0; mt < 2; ++mt)
#pragma unroll
        for (int u2 = 0; u2 < 2; ++u2) {
            v8f vi = acc[mt][0][u2], vf = acc[mt][1][u2], vg = acc[mt][2][u2], vo = acc[mt][3][u2];
#pragma unroll
            for (int r = 0; r < 8; ++r) {
                int row = mt * 16 + r + ((lane & 16) ? 8 : 0);
                int cu = ub + u2 * 16 + (lane & 15);
                const float* op = obs_part + (size_t)(b0 + row) * G4 + cu;
                float ti = 0.f, tf = 0.f, tg = 0.f, to = 0.f;
                if (tab) {
                    const float* tr = tab + (size_t)selL[row] * G4 + cu;
                    ti = tr[0]; tf = tr[256]; tg = tr[512]; to = tr[768];
                }
                float gi = vi[r] + op[0]   + ab[cu]       + ti;
                float gf = vf[r] + op[256] + ab[256 + cu] + tf;
                float gg = vg[r] + op[512] + ab[512 + cu] + tg;
                float go = vo[r] + op[768] + ab[768 + cu] + to;
                float cn = sigmoidf_(gf) * creg[mt][u2][r] + sigmoidf_(gi) * tanhf(gg);
                creg[mt][u2][r] = cn;
                float hv = sigmoidf_(go) * tanhf(cn);
                hL[row * 256 + cu] = (_Float16)hv;
            }
        }
    __syncthreads();   // hL(new) visible to everyone
}

__device__ __forceinline__ float dot_h256(const _Float16* hrow, const float* wrow) {
    const h8* hr = (const h8*)hrow;
    const f4v* wv = (const f4v*)wrow;
    float s = 0.f;
#pragma unroll 4
    for (int kb = 0; kb < 32; ++kb) {
        h8 hv = hr[kb];
        f4v w0 = wv[2 * kb], w1 = wv[2 * kb + 1];
        s += (float)hv[0] * w0[0] + (float)hv[1] * w0[1] + (float)hv[2] * w0[2] + (float)hv[3] * w0[3]
           + (float)hv[4] * w1[0] + (float)hv[5] * w1[1] + (float)hv[6] * w1[2] + (float)hv[7] * w1[3];
    }
    return s;
}

__device__ __forceinline__ void head_cat(const _Float16* hL, float* yL,
                                         const float* __restrict__ W, const float* __restrict__ bias,
                                         const int* selL, float* lp, int tid) {
    if (tid < 64) {
        int row = tid >> 1, j = tid & 1;
        yL[row * 4 + j] = bias[j] + dot_h256(hL + row * 256, W + j * HID);
    }
    __syncthreads();
    if (tid < 32) {
        float y0 = yL[tid * 4], y1 = yL[tid * 4 + 1];
        float m = fmaxf(y0, y1);
        float lse = m + logf(expf(y0 - m) + expf(y1 - m));
        lp[tid] = (selL[tid] ? y1 : y0) - lse;
    }
    __syncthreads();
}

__device__ __forceinline__ void head_norm(const _Float16* hL, float* yL,
                                          const float* __restrict__ Wr, const float* __restrict__ br,
                                          const float* __restrict__ raw_pos, int pt, int n0,
                                          float* lp, int tid) {
    if (tid < 128) {
        int row = tid >> 2, j = tid & 3;
        yL[row * 4 + j] = br[j] + dot_h256(hL + row * 256, Wr + j * HID);
    }
    __syncthreads();
    if (tid < 32) {
        int n = n0 + tid;
        float x0 = raw_pos[(size_t)n * 4 + pt * 2 + 0];
        float x1 = raw_pos[(size_t)n * 4 + pt * 2 + 1];
        float l0 = yL[tid * 4], l1 = yL[tid * 4 + 1];
        float s0 = yL[tid * 4 + 2], s1 = yL[tid * 4 + 3];
        float z0 = (x0 - l0) * expf(-s0);
        float z1 = (x1 - l1) * expf(-s1);
        lp[tid] = -0.5f * (z0 * z0 + z1 * z1) - (s0 + s1) - LOG2PI_F;
    }
    __syncthreads();
}

// ---------------------------------------------------------------------------
// Kernel 5: main recurrence. 32 rows / workgroup (consecutive b, same s).
// ---------------------------------------------------------------------------
__launch_bounds__(256, 1)
__global__ void k_main(const _Float16* __restrict__ whh_h, const float* __restrict__ obs_part,
                       const float* __restrict__ addr_bias, const float* __restrict__ tab_prog,
                       const float* __restrict__ tab_shape,
                       const _Float16* __restrict__ wihmid, const _Float16* __restrict__ rpe,
                       const _Float16* __restrict__ h1h, const float* __restrict__ c1,
                       const float* __restrict__ plog,
                       const int* __restrict__ program_id, const int* __restrict__ shape_ids,
                       const float* __restrict__ raw_pos,
                       const float* __restrict__ Ws, const float* __restrict__ bs,
                       const float* __restrict__ Wr, const float* __restrict__ br,
                       float* __restrict__ out) {
    __shared__ __align__(16) _Float16 hL[32 * 256];
    __shared__ float yL[32 * 4];
    __shared__ int pidL[32], s0L[32], s1L[32];
    __shared__ float lpA[32], lpB[32], lpC[32], lpD[32], lpE[32], lpF[32];

    int tid = threadIdx.x, wave = tid >> 5, lane = tid & 31;
    int blk = blockIdx.x;
    int s = blk >> 7;
    int b0 = (blk & 127) << 5;
    int n0 = s * BB + b0;

    if (tid < 32) {
        int n = n0 + tid;
        pidL[tid] = program_id[n];
        s0L[tid] = shape_ids[2 * n];
        s1L[tid] = shape_ids[2 * n + 1];
    }
    // stage h1 for steps A..B (16B vectorized copy: 32*256 f16 = 1024 uint4)
    {
        const uint4* src = (const uint4*)(h1h + (size_t)b0 * 256);
        uint4* dst = (uint4*)hL;
        for (int i = tid; i < 1024; i += 256) dst[i] = src[i];
    }
    __syncthreads();

    v8f creg[2][2];

    // step A: addr 1, prev = pid emb, state (h1,c1)   -> lp_shape
    do_step(hL, creg, whh_h, obs_part, addr_bias + 1 * G4, tab_prog, pidL, nullptr, wihmid, c1, true, b0, n0, wave, lane);
    head_cat(hL, yL, Ws, bs, s0L, lpA, tid);
    // step B: addr 4, prev = sid0 emb                 -> lp_rp (rpos0)
    do_step(hL, creg, whh_h, obs_part, addr_bias + 4 * G4, tab_shape, s0L, nullptr, wihmid, c1, false, b0, n0, wave, lane);
    head_norm(hL, yL, Wr, br, raw_pos, 0, n0, lpB, tid);

    // restart from (h1,c1) for branch 1
    {
        const uint4* src = (const uint4*)(h1h + (size_t)b0 * 256);
        uint4* dst = (uint4*)hL;
        for (int i = tid; i < 1024; i += 256) dst[i] = src[i];
    }
    __syncthreads();

    // step C: addr 2, prev = pid emb                  -> lp_s0
    do_step(hL, creg, whh_h, obs_part, addr_bias + 2 * G4, tab_prog, pidL, nullptr, wihmid, c1, true, b0, n0, wave, lane);
    head_cat(hL, yL, Ws, bs, s0L, lpC, tid);
    // step D: addr 3, prev = sid0 emb                 -> lp_s1
    do_step(hL, creg, whh_h, obs_part, addr_bias + 3 * G4, tab_shape, s0L, nullptr, wihmid, c1, false, b0, n0, wave, lane);
    head_cat(hL, yL, Ws, bs, s1L, lpD, tid);
    // step E: addr 5, prev = sid1 emb                 -> lp_r0 (rpos0)
    do_step(hL, creg, whh_h, obs_part, addr_bias + 5 * G4, tab_shape, s1L, nullptr, wihmid, c1, false, b0, n0, wave, lane);
    head_norm(hL, yL, Wr, br, raw_pos, 0, n0, lpE, tid);
    // step F: addr 6, prev = rp_emb (via extra WMMA K-tile) -> lp_r1 (rpos1)
    do_step(hL, creg, whh_h, obs_part, addr_bias + 6 * G4, nullptr, nullptr, rpe, wihmid, c1, false, b0, n0, wave, lane);
    head_norm(hL, yL, Wr, br, raw_pos, 1, n0, lpF, tid);

    if (tid < 32) {
        int n = n0 + tid;
        int pid = pidL[tid];
        float pl = plog[(b0 + tid) * 3 + pid];
        float r = (pid == 0) ? (pl + lpA[tid] + lpB[tid])
                             : (pl + lpC[tid] + lpD[tid] + lpE[tid] + lpF[tid]);
        out[n] = r;
    }
}

// ---------------------------------------------------------------------------
// Launcher
// ---------------------------------------------------------------------------
extern "C" void kernel_launch(void* const* d_in, const int* in_sizes, int n_in,
                              void* d_out, int out_size, void* d_ws, size_t ws_size,
                              hipStream_t stream) {
    const float* obs      = (const float*)d_in[0];   // (B,400)
    const float* raw_pos  = (const float*)d_in[1];   // (S,B,2,2)
    const int*   prog_id  = (const int*)d_in[2];     // (S,B)
    const int*   shape_id = (const int*)d_in[3];     // (S,B,2)
    const float* addr_emb = (const float*)d_in[4];   // (7,16)
    const float* pid_emb  = (const float*)d_in[5];   // (3,16)
    const float* sid_emb  = (const float*)d_in[6];   // (2,16)
    const float* W_ih     = (const float*)d_in[7];   // (1024,432)
    const float* W_hh     = (const float*)d_in[8];   // (1024,256)
    const float* b_ih     = (const float*)d_in[9];
    const float* b_hh     = (const float*)d_in[10];
    const float* mW1      = (const float*)d_in[11];
    const float* mb1      = (const float*)d_in[12];
    const float* mW2      = (const float*)d_in[13];
    const float* mb2      = (const float*)d_in[14];
    const float* mW3      = (const float*)d_in[15];
    const float* mb3      = (const float*)d_in[16];
    const float* Wp       = (const float*)d_in[17];
    const float* bp       = (const float*)d_in[18];
    const float* Ws       = (const float*)d_in[19];
    const float* bs       = (const float*)d_in[20];
    const float* Wr       = (const float*)d_in[21];
    const float* br       = (const float*)d_in[22];
    float* out = (float*)d_out;

    // workspace carve-up (all offsets 64B-aligned)
    char* ws = (char*)d_ws;
    size_t off = 0;
    float*    obs_part = (float*)(ws + off);    off += (size_t)BB * G4 * 4;       // 16 MB
    _Float16* whh_h    = (_Float16*)(ws + off); off += (size_t)G4 * HID * 2;      // 512 KB
    _Float16* h1h      = (_Float16*)(ws + off); off += (size_t)BB * HID * 2;      // 2 MB
    float*    c1buf    = (float*)(ws + off);    off += (size_t)BB * HID * 4;      // 4 MB
    float*    plog     = (float*)(ws + off);    off += (size_t)BB * 3 * 4 + 192;  // ~48 KB
    float*    addr_b   = (float*)(ws + off);    off += (size_t)7 * G4 * 4;
    float*    tab_p    = (float*)(ws + off);    off += (size_t)3 * G4 * 4;
    float*    tab_s    = (float*)(ws + off);    off += (size_t)2 * G4 * 4;
    _Float16* wihmid   = (_Float16*)(ws + off); off += (size_t)G4 * 16 * 2;
    _Float16* rpe      = (_Float16*)(ws + off); off += (size_t)NTOT * 16 * 2;     // 2 MB
    (void)ws_size; (void)in_sizes; (void)n_in; (void)out_size;

    k_tables<<<G4, 256, 0, stream>>>(W_ih, W_hh, b_ih, b_hh, addr_emb, pid_emb, sid_emb,
                                     addr_b, tab_p, tab_s, whh_h, wihmid);
    k_obs_gemm<<<dim3(BB / 128, G4 / 128), 256, 0, stream>>>(obs, W_ih, obs_part);
    k_h1<<<BB, 256, 0, stream>>>(obs_part, addr_b, Wp, bp, c1buf, h1h, plog);
    k_mlp<<<NTOT, 128, 0, stream>>>(raw_pos, mW1, mb1, mW2, mb2, mW3, mb3, rpe);
    k_main<<<NTOT / 32, 256, 0, stream>>>(whh_h, obs_part, addr_b, tab_p, tab_s, wihmid, rpe,
                                          h1h, c1buf, plog, prog_id, shape_id, raw_pos,
                                          Ws, bs, Wr, br, out);
}
